// VanillaLlamaAttention_31301721653896
// MI455X (gfx1250) — compile-verified, matrix operand order checked by reading, not run on hardware
//
#include <hip/hip_runtime.h>

// Problem constants (B=1)
#define S_LEN   2048
#define NH      32
#define NKV     8
#define HD      64
#define HID_DIM 2048
#define KIN     4096   // 2*HID
#define NQKV    3072   // NH*HD + 2*NKV*HD

typedef __attribute__((ext_vector_type(16))) __bf16       v16bf;
typedef __attribute__((ext_vector_type(8)))  float        v8f;
typedef __attribute__((ext_vector_type(4)))  unsigned int v4u;

union FragBF { v16bf v; v4u q[2]; };

// CDNA5 async global->LDS path (ASYNCcnt-tracked DMA, no VGPR round trip).
// Probe (round 2 stderr): param 0 is `int __attribute__((vector_size(16)))`
// in the global (__device__/AS1) address space; dest is the AS3 twin.
#if __has_builtin(__builtin_amdgcn_global_load_async_to_lds_b128) && \
    __has_builtin(__builtin_amdgcn_s_wait_asynccnt)
#define HAVE_ASYNC_LDS 1
typedef int v4i_vec __attribute__((vector_size(16)));
typedef __attribute__((address_space(1))) v4i_vec global_v4i;
typedef __attribute__((address_space(3))) v4i_vec lds_v4i;
#else
#define HAVE_ASYNC_LDS 0
#endif

// ---------------------------------------------------------------------------
// fp32 -> bf16 conversion (bandwidth-bound, 4-wide)
// ---------------------------------------------------------------------------
__global__ void cvt_f32_bf16(const float* __restrict__ src,
                             __bf16* __restrict__ dst, int n4) {
  int i = blockIdx.x * blockDim.x + threadIdx.x;
  if (i >= n4) return;
  float4 f = reinterpret_cast<const float4*>(src)[i];
  dst[i * 4 + 0] = (__bf16)f.x;
  dst[i * 4 + 1] = (__bf16)f.y;
  dst[i * 4 + 2] = (__bf16)f.z;
  dst[i * 4 + 3] = (__bf16)f.w;
}

// ---------------------------------------------------------------------------
// C(MxN, f32) = A(MxK, bf16 row-major) @ W(NxK, bf16 row-major)^T
// 256 threads = 8 waves; each wave computes a 32x64 tile (8 WMMAs / K-step);
// block covers 256x64.  W tile (64x32) double-buffered in LDS via async DMA.
// ---------------------------------------------------------------------------
__global__ __launch_bounds__(256)
void gemm_bf16_nt(const __bf16* __restrict__ A, const __bf16* __restrict__ W,
                  float* __restrict__ C, int M, int N, int K) {
  __shared__ __align__(16) __bf16 wtile[2][64 * 32];
  const int wave = threadIdx.x >> 5;
  const int lane = threadIdx.x & 31;
  const int g = lane >> 4;  // 16-lane group within wave
  const int ln = lane & 15;
  const int m0 = blockIdx.x * 256 + wave * 32;
  const int n0 = blockIdx.y * 64;

  v8f acc[2][4] = {};
  const __bf16* arow0 = A + (size_t)(m0 + ln) * K;
  const __bf16* arow1 = A + (size_t)(m0 + 16 + ln) * K;

  // each thread stages one 16B chunk of the 64x32 W tile
  const int sr = threadIdx.x >> 2;         // tile row 0..63
  const int sc = (threadIdx.x & 3) << 3;   // tile col 0,8,16,24
  const __bf16* wsrc = &W[(size_t)(n0 + sr) * K + sc];
  __bf16* wdst0 = &wtile[0][sr * 32 + sc];
  __bf16* wdst1 = &wtile[1][sr * 32 + sc];
  const int niter = K / 32;

#if HAVE_ASYNC_LDS
  __builtin_amdgcn_global_load_async_to_lds_b128((global_v4i*)wsrc,
                                                 (lds_v4i*)wdst0, 0, 0);
#endif

  for (int i = 0; i < niter; ++i) {
    const int k0 = i * 32;
    const int cur = i & 1;
#if HAVE_ASYNC_LDS
    if (i + 1 < niter) {  // prefetch next tile into the other buffer
      __builtin_amdgcn_global_load_async_to_lds_b128(
          (global_v4i*)(wsrc + k0 + 32), (lds_v4i*)(cur ? wdst0 : wdst1), 0, 0);
      __builtin_amdgcn_s_wait_asynccnt(1);  // current tile landed
    } else {
      __builtin_amdgcn_s_wait_asynccnt(0);
    }
#else
    *reinterpret_cast<v4u*>(cur ? wdst1 : wdst0) =
        *reinterpret_cast<const v4u*>(wsrc + k0);
#endif
    __syncthreads();  // whole tile visible to all waves

    FragBF a0, a1;
    a0.q[0] = *reinterpret_cast<const v4u*>(arow0 + k0 + 8 * g);
    a0.q[1] = *reinterpret_cast<const v4u*>(arow0 + k0 + 16 + 8 * g);
    a1.q[0] = *reinterpret_cast<const v4u*>(arow1 + k0 + 8 * g);
    a1.q[1] = *reinterpret_cast<const v4u*>(arow1 + k0 + 16 + 8 * g);
    if (k0 + 32 < K) __builtin_prefetch(arow0 + k0 + 32, 0, 0);

#pragma unroll
    for (int t = 0; t < 4; ++t) {
      FragBF b;
      const __bf16* wr = &wtile[cur][(t * 16 + ln) * 32];
      b.q[0] = *reinterpret_cast<const v4u*>(wr + 8 * g);
      b.q[1] = *reinterpret_cast<const v4u*>(wr + 16 + 8 * g);
      acc[0][t] = __builtin_amdgcn_wmma_f32_16x16x32_bf16(
          false, a0.v, false, b.v, (short)0, acc[0][t], false, false);
      acc[1][t] = __builtin_amdgcn_wmma_f32_16x16x32_bf16(
          false, a1.v, false, b.v, (short)0, acc[1][t], false, false);
    }
    __syncthreads();  // done reading before next async overwrite
  }

#pragma unroll
  for (int hh = 0; hh < 2; ++hh)
#pragma unroll
    for (int t = 0; t < 4; ++t)
#pragma unroll
      for (int r = 0; r < 8; ++r)
        C[(size_t)(m0 + hh * 16 + r + 8 * g) * N + n0 + t * 16 + ln] =
            acc[hh][t][r];
}

// ---------------------------------------------------------------------------
// RoPE + relayout.  QKVf is [S][3072] fp32 (q|k|v columns).
// Q gets D^-1/2 folded in.  Q/K head-major [h][s][d]; V TRANSPOSED [kvh][d][s]
// so the attention PV B-fragments are contiguous along the KV axis.
// ---------------------------------------------------------------------------
__global__ void rope_layout(const float* __restrict__ qkv,
                            const int* __restrict__ pos_ids,
                            __bf16* __restrict__ Qh, __bf16* __restrict__ Kh,
                            __bf16* __restrict__ Vt) {
  int idx = blockIdx.x * blockDim.x + threadIdx.x;
  if (idx >= S_LEN * NQKV) return;
  int s = idx / NQKV;
  int c = idx - s * NQKV;
  float x = qkv[idx];
  if (c < NH * HD + NKV * HD) {
    bool isq = (c < NH * HD);
    int cc = isq ? c : (c - NH * HD);
    int hh = cc >> 6, d = cc & 63;
    float pos = (float)pos_ids[s];
    int i = d & 31;
    float invf = powf(10000.0f, -(float)(2 * i) * (1.0f / 64.0f));
    float sn, cs;
    __sincosf(pos * invf, &sn, &cs);
    int base = s * NQKV + (isq ? 0 : NH * HD) + (hh << 6);
    float other = qkv[base + ((d < 32) ? d + 32 : d - 32)];
    float rot = (d < 32) ? -other : other;
    float val = x * cs + rot * sn;
    if (isq)
      Qh[((size_t)hh * S_LEN + s) * HD + d] = (__bf16)(val * 0.125f);
    else
      Kh[((size_t)hh * S_LEN + s) * HD + d] = (__bf16)val;
  } else {
    int cc = c - NH * HD - NKV * HD;
    int hh = cc >> 6, d = cc & 63;
    Vt[((size_t)hh * HD + d) * S_LEN + s] = (__bf16)x;  // transposed
  }
}

// ---------------------------------------------------------------------------
// Flash attention (causal, GQA 4:1). Grid (S/128, NH), 256 threads = 8 waves.
// Each wave: 16 query rows x D=64 accumulator; KV consumed 32 cols/step.
// Scores: 4 WMMAs;  P@V: 4 WMMAs from V^T (b128 loads, no V staging).
// P transposed C-layout -> A-layout via per-wave LDS bounce.
// ---------------------------------------------------------------------------
__global__ __launch_bounds__(256)
void attn_kernel(const __bf16* __restrict__ Qh, const __bf16* __restrict__ Kh,
                 const __bf16* __restrict__ Vt, __bf16* __restrict__ AO) {
  __shared__ __align__(16) __bf16 ps[8][16 * 32];  // P transpose bounce

  const int wave = threadIdx.x >> 5;
  const int lane = threadIdx.x & 31;
  const int g = lane >> 4;
  const int ln = lane & 15;
  const int h = blockIdx.y;
  const int kvh = h >> 2;  // H/HK = 4
  const int qbase = blockIdx.x * 128 + wave * 16;
  __bf16* psw = &ps[wave][0];

  // Preload Q fragments for k-slices [0,32) and [32,64)
  FragBF aq0, aq1;
  const __bf16* qrow = Qh + ((size_t)h * S_LEN + qbase + ln) * HD;
  aq0.q[0] = *reinterpret_cast<const v4u*>(qrow + 8 * g);
  aq0.q[1] = *reinterpret_cast<const v4u*>(qrow + 16 + 8 * g);
  aq1.q[0] = *reinterpret_cast<const v4u*>(qrow + 32 + 8 * g);
  aq1.q[1] = *reinterpret_cast<const v4u*>(qrow + 48 + 8 * g);

  // V^T row for this lane's output column in each 16-wide tile
  const __bf16* vtrow[4];
#pragma unroll
  for (int t = 0; t < 4; ++t)
    vtrow[t] = Vt + ((size_t)kvh * HD + t * 16 + ln) * S_LEN;

  float mrow[8], lrow[8];
  v8f acc[4] = {};
#pragma unroll
  for (int r = 0; r < 8; ++r) { mrow[r] = -1e30f; lrow[r] = 0.0f; }

  const int kvend = qbase + 16;  // causal bound
  for (int kv0 = 0; kv0 < kvend; kv0 += 32) {
    // Scores: two 16x16 tiles, each = 2 WMMAs over D=64
    v8f sc[2];
#pragma unroll
    for (int t2 = 0; t2 < 2; ++t2) {
      const __bf16* krow = Kh + ((size_t)kvh * S_LEN + kv0 + t2 * 16 + ln) * HD;
      FragBF bk0, bk1;
      bk0.q[0] = *reinterpret_cast<const v4u*>(krow + 8 * g);
      bk0.q[1] = *reinterpret_cast<const v4u*>(krow + 16 + 8 * g);
      bk1.q[0] = *reinterpret_cast<const v4u*>(krow + 32 + 8 * g);
      bk1.q[1] = *reinterpret_cast<const v4u*>(krow + 48 + 8 * g);
      v8f s = {};
      s = __builtin_amdgcn_wmma_f32_16x16x32_bf16(false, aq0.v, false, bk0.v,
                                                  (short)0, s, false, false);
      s = __builtin_amdgcn_wmma_f32_16x16x32_bf16(false, aq1.v, false, bk1.v,
                                                  (short)0, s, false, false);
      sc[t2] = s;
    }

    // Causal mask + online softmax (row stats via 16-lane butterflies)
    float scal[8];
#pragma unroll
    for (int r = 0; r < 8; ++r) {
      const int row = qbase + r + 8 * g;
#pragma unroll
      for (int t2 = 0; t2 < 2; ++t2) {
        const int col = kv0 + t2 * 16 + ln;
        if (col > row) sc[t2][r] = -1e30f;
      }
      float mx = fmaxf(sc[0][r], sc[1][r]);
#pragma unroll
      for (int o = 1; o < 16; o <<= 1) mx = fmaxf(mx, __shfl_xor(mx, o, 32));
      const float mnew = fmaxf(mrow[r], mx);
      scal[r] = __expf(mrow[r] - mnew);
      const float p0 = __expf(sc[0][r] - mnew);
      const float p1 = __expf(sc[1][r] - mnew);
      sc[0][r] = p0;
      sc[1][r] = p1;
      float psum = p0 + p1;
#pragma unroll
      for (int o = 1; o < 16; o <<= 1) psum += __shfl_xor(psum, o, 32);
      lrow[r] = lrow[r] * scal[r] + psum;
      mrow[r] = mnew;
    }
#pragma unroll
    for (int t = 0; t < 4; ++t)
#pragma unroll
      for (int r = 0; r < 8; ++r) acc[t][r] *= scal[r];

    // P: C-layout -> row-major 16x32 in LDS, re-read as A fragment
#pragma unroll
    for (int t2 = 0; t2 < 2; ++t2)
#pragma unroll
      for (int r = 0; r < 8; ++r)
        psw[(r + 8 * g) * 32 + t2 * 16 + ln] = (__bf16)sc[t2][r];
    FragBF ap;
    ap.q[0] = *reinterpret_cast<const v4u*>(psw + ln * 32 + 8 * g);
    ap.q[1] = *reinterpret_cast<const v4u*>(psw + ln * 32 + 16 + 8 * g);

    // acc += P(16x32) @ V(32x64): B frags straight from V^T (contiguous in K)
#pragma unroll
    for (int t = 0; t < 4; ++t) {
      FragBF bv;
      bv.q[0] = *reinterpret_cast<const v4u*>(vtrow[t] + kv0 + 8 * g);
      bv.q[1] = *reinterpret_cast<const v4u*>(vtrow[t] + kv0 + 16 + 8 * g);
      acc[t] = __builtin_amdgcn_wmma_f32_16x16x32_bf16(
          false, ap.v, false, bv.v, (short)0, acc[t], false, false);
    }
  }

  // Epilogue: normalize, write [s][h*64+d] bf16 for the output projection
#pragma unroll
  for (int t = 0; t < 4; ++t)
#pragma unroll
    for (int r = 0; r < 8; ++r) {
      const int row = qbase + r + 8 * g;
      AO[(size_t)row * (NH * HD) + h * HD + t * 16 + ln] =
          (__bf16)(acc[t][r] / lrow[r]);
    }
}

// ---------------------------------------------------------------------------
extern "C" void kernel_launch(void* const* d_in, const int* in_sizes, int n_in,
                              void* d_out, int out_size, void* d_ws,
                              size_t ws_size, hipStream_t stream) {
  const float* hs = (const float*)d_in[0];
  const int* pos = (const int*)d_in[1];
  const float* wq = (const float*)d_in[2];
  const float* wk = (const float*)d_in[3];
  const float* wv = (const float*)d_in[4];
  const float* wo = (const float*)d_in[5];
  (void)in_sizes; (void)n_in; (void)out_size; (void)ws_size;

  char* ws = (char*)d_ws;
  size_t off = 0;
  auto carve = [&](size_t bytes) {
    char* p = ws + off;
    off = (off + bytes + 255) & ~(size_t)255;
    return p;
  };
  __bf16* Xb    = (__bf16*)carve((size_t)S_LEN * KIN * 2);
  __bf16* Wqkvb = (__bf16*)carve((size_t)NQKV * KIN * 2);
  __bf16* Wob   = (__bf16*)carve((size_t)HID_DIM * NH * HD * 2);
  float*  QKVf  = (float*) carve((size_t)S_LEN * NQKV * 4);
  __bf16* Qh    = (__bf16*)carve((size_t)NH * S_LEN * HD * 2);
  __bf16* Kh    = (__bf16*)carve((size_t)NKV * S_LEN * HD * 2);
  __bf16* Vt    = (__bf16*)carve((size_t)NKV * S_LEN * HD * 2);
  __bf16* AO    = (__bf16*)carve((size_t)S_LEN * NH * HD * 2);

  auto cvt = [&](const float* src, __bf16* dst, size_t n) {
    int n4 = (int)(n / 4);
    cvt_f32_bf16<<<(n4 + 255) / 256, 256, 0, stream>>>(src, dst, n4);
  };
  cvt(hs, Xb, (size_t)S_LEN * KIN);
  cvt(wq, Wqkvb, (size_t)NH * HD * KIN);
  cvt(wk, Wqkvb + (size_t)NH * HD * KIN, (size_t)NKV * HD * KIN);
  cvt(wv, Wqkvb + (size_t)(NH * HD + NKV * HD) * KIN, (size_t)NKV * HD * KIN);
  cvt(wo, Wob, (size_t)HID_DIM * NH * HD);

  // Fused QKV projection: [2048 x 4096] @ [3072 x 4096]^T
  gemm_bf16_nt<<<dim3(S_LEN / 256, NQKV / 64), 256, 0, stream>>>(
      Xb, Wqkvb, QKVf, S_LEN, NQKV, KIN);

  rope_layout<<<(S_LEN * NQKV + 255) / 256, 256, 0, stream>>>(QKVf, pos, Qh,
                                                              Kh, Vt);

  attn_kernel<<<dim3(S_LEN / 128, NH), 256, 0, stream>>>(Qh, Kh, Vt, AO);

  // Output projection: [2048 x 2048] @ [2048 x 2048]^T -> fp32 d_out
  gemm_bf16_nt<<<dim3(S_LEN / 256, HID_DIM / 64), 256, 0, stream>>>(
      AO, Wob, (float*)d_out, S_LEN, HID_DIM, NH * HD);
}
